// LSTM_51866025066917
// MI455X (gfx1250) — matrix operand order repaired and checked by reading, not compiled, not run on hardware
//
#include <hip/hip_runtime.h>

// ---------------------------------------------------------------------------
// Types for CDNA5 WMMA
// ---------------------------------------------------------------------------
typedef __attribute__((ext_vector_type(16))) __bf16 v16bf;
typedef __attribute__((ext_vector_type(8)))  __bf16 v8bf;
typedef __attribute__((ext_vector_type(8)))  float  v8f;

// Problem constants
#define BB   64
#define TT   512
#define DD   256
#define UU   1024
#define MM   960
#define U4   (4 * UU)

// ---------------------------------------------------------------------------
// Helpers
// ---------------------------------------------------------------------------
__device__ __forceinline__ unsigned short f32_to_bf16(float f) {
    union { float f; unsigned u; } x; x.f = f;
    unsigned r = x.u + 0x7FFFu + ((x.u >> 16) & 1u);   // round-to-nearest-even
    return (unsigned short)(r >> 16);
}

__device__ __forceinline__ float fast_sigmoid(float x) {
    return 1.0f / (1.0f + __expf(-x));
}
__device__ __forceinline__ float fast_tanh(float x) {
    return 2.0f / (1.0f + __expf(-2.0f * x)) - 1.0f;
}

// A fragment: 16x32 bf16 tile, row-major source with leading dim ld.
// Lane L<16: row L,   K in {k0..k0+7} U {k0+16..k0+23}
// Lane L>=16: row L-16, K in {k0+8..k0+15} U {k0+24..k0+31}
__device__ __forceinline__ v16bf load_frag_a(const unsigned short* base, int ld,
                                             int k0, int lane) {
    const int row  = lane & 15;
    const int koff = (lane >> 4) << 3;
    const unsigned short* p = base + (size_t)row * ld + k0 + koff;
    union { v16bf v; v8bf h[2]; } f;
    f.h[0] = *reinterpret_cast<const v8bf*>(p);
    f.h[1] = *reinterpret_cast<const v8bf*>(p + 16);
    return f.v;
}

// B fragment: 32x16 bf16 tile from an N-major (pre-transposed) matrix
// Bt[N][K] with leading dim ld.  Lane L<16: col L, K in {k0..k0+7} U {k0+16..k0+23};
// lane L>=16: col L-16, K in {k0+8..k0+15} U {k0+24..k0+31}.  Two b128 loads.
__device__ __forceinline__ v16bf load_frag_b(const unsigned short* base, int ld,
                                             int k0, int lane) {
    const int col = lane & 15;
    const int kb  = (lane >> 4) << 3;
    const unsigned short* p = base + (size_t)col * ld + k0 + kb;
    union { v16bf v; v8bf h[2]; } f;
    f.h[0] = *reinterpret_cast<const v8bf*>(p);
    f.h[1] = *reinterpret_cast<const v8bf*>(p + 16);
    return f.v;
}

__device__ __forceinline__ v8f wmma_bf16(v16bf a, v16bf b, v8f c) {
    return __builtin_amdgcn_wmma_f32_16x16x32_bf16(
        /*neg_a=*/false, a, /*neg_b=*/false, b,
        /*c_mod=*/(short)0, c, /*reuse_a=*/false, /*reuse_b=*/false);
}

// ---------------------------------------------------------------------------
// Conversion / init kernels
// ---------------------------------------------------------------------------
__global__ void k_convert_bf16(const float* __restrict__ src,
                               unsigned short* __restrict__ dst, int n) {
    int i = blockIdx.x * blockDim.x + threadIdx.x;
    if (i < n) dst[i] = f32_to_bf16(src[i]);
}

// src [R,C] f32 row-major -> dst [C,R] bf16 (N-major for B fragments)
__global__ void k_transpose_bf16(const float* __restrict__ src,
                                 unsigned short* __restrict__ dst,
                                 int R, int C) {
    int i = blockIdx.x * blockDim.x + threadIdx.x;
    if (i < R * C) {
        int r = i / C, c = i - r * C;
        dst[(size_t)c * R + r] = f32_to_bf16(src[i]);
    }
}

__global__ void k_init_state(const float* __restrict__ h0,
                             const float* __restrict__ c0,
                             unsigned short* __restrict__ h_bf,
                             float* __restrict__ c_ws, int n) {
    int i = blockIdx.x * blockDim.x + threadIdx.x;
    if (i < n) { h_bf[i] = f32_to_bf16(h0[i]); c_ws[i] = c0[i]; }
}

// ---------------------------------------------------------------------------
// Generic WMMA GEMM:  C[M,N] f32 = A[M,K]bf16 (row-major) x Bt[N,K]bf16 + bias
// block = 256 threads (8 waves).  Wave computes a 32x64 tile (2x4 subtiles);
// waves stack along M -> workgroup tile = 256 rows x 64 cols.
// grid = (N/64, M/256).  K-loop is software-pipelined (double-buffered
// fragments) so global loads for step k+1 overlap the 8 WMMAs of step k.
// ---------------------------------------------------------------------------
struct GemmFrags { v16bf a[2]; v16bf b[4]; };

__device__ __forceinline__ void gemm_load_tile(GemmFrags& f,
                                               const unsigned short* Arow,
                                               const unsigned short* Bt,
                                               int K, int n0, int k0, int lane) {
    f.a[0] = load_frag_a(Arow, K, k0, lane);
    f.a[1] = load_frag_a(Arow + (size_t)16 * K, K, k0, lane);
#pragma unroll
    for (int j = 0; j < 4; ++j)
        f.b[j] = load_frag_b(Bt + (size_t)(n0 + j * 16) * K, K, k0, lane);
}

__global__ void k_gemm_bf16(const unsigned short* __restrict__ A,
                            const unsigned short* __restrict__ Bt,
                            const float* __restrict__ bias,
                            float* __restrict__ C,
                            int M, int N, int K) {
    const int wave = threadIdx.x >> 5;
    const int lane = threadIdx.x & 31;
    const int m0 = (blockIdx.y * 8 + wave) * 32;
    const int n0 = blockIdx.x * 64;
    if (m0 >= M) return;

    v8f acc[2][4] = {};
    const unsigned short* Arow = A + (size_t)m0 * K;

    GemmFrags cur, nxt;
    gemm_load_tile(cur, Arow, Bt, K, n0, 0, lane);

    for (int k0 = 32; k0 < K; k0 += 32) {
        gemm_load_tile(nxt, Arow, Bt, K, n0, k0, lane);   // overlap with WMMA
#pragma unroll
        for (int mi = 0; mi < 2; ++mi)
#pragma unroll
            for (int j = 0; j < 4; ++j)
                acc[mi][j] = wmma_bf16(cur.a[mi], cur.b[j], acc[mi][j]);
        cur = nxt;
    }
#pragma unroll
    for (int mi = 0; mi < 2; ++mi)
#pragma unroll
        for (int j = 0; j < 4; ++j)
            acc[mi][j] = wmma_bf16(cur.a[mi], cur.b[j], acc[mi][j]);

    const int col = lane & 15, hi = lane >> 4;
#pragma unroll
    for (int mi = 0; mi < 2; ++mi)
#pragma unroll
        for (int j = 0; j < 4; ++j) {
            const int n = n0 + j * 16 + col;
            const float bb = bias ? bias[n] : 0.0f;
#pragma unroll
            for (int v = 0; v < 8; ++v) {
                const int m = m0 + mi * 16 + hi * 8 + v;
                C[(size_t)m * N + n] = acc[mi][j][v] + bb;
            }
        }
}

// ---------------------------------------------------------------------------
// One LSTM timestep, fully fused:
//   z = xz[:,t,:] + h_prev @ R ; gates ; c,h update ; h -> bf16 ping-pong +
//   bf16 lstm_out for the final GEMM.
// block = 128 threads (4 waves), grid = (U/16, B/16).
//
// The 16x1024 bf16 h tile (contiguous 32 KB) is staged into LDS with
// GLOBAL_LOAD_ASYNC_TO_LDS_B128 (ASYNCcnt path), then A-fragments are read
// from LDS while the 4 waves split K (256 each) with double-buffered Rt
// fragments.  Partial sums are reduced through LDS and the gate epilogue is
// spread across all 4 waves.
// ---------------------------------------------------------------------------
struct StepFrags { v16bf a; v16bf b[4]; };

__device__ __forceinline__ void step_load_tile(StepFrags& f,
                                               const unsigned short* hsm,
                                               const unsigned short* Rt,
                                               int u0, int k0, int lane) {
    f.a = load_frag_a(hsm, UU, k0, lane);                 // LDS reads
#pragma unroll
    for (int g = 0; g < 4; ++g)
        f.b[g] = load_frag_b(Rt + (size_t)(g * UU + u0) * UU, UU, k0, lane);
}

__global__ void k_lstm_step(const unsigned short* __restrict__ h_prev, // [B,U] bf16
                            const unsigned short* __restrict__ Rt,     // [4U,U] bf16
                            const float* __restrict__ xz,              // [B,T,4U]
                            float* __restrict__ c_state,               // [B,U]
                            unsigned short* __restrict__ h_next,       // [B,U] bf16
                            unsigned short* __restrict__ lstm_bf,      // [B,T,U] bf16
                            float* __restrict__ h_last,                // [B,U] f32
                            float* __restrict__ c_last,                // [B,U] f32
                            int t, int is_last) {
    __shared__ unsigned short hsm[16 * UU];  // 32 KB staged h tile
    __shared__ float part[4 * 4 * 8 * 32];   // [wave][gate][v][lane], 16 KB

    const int wave = threadIdx.x >> 5;
    const int lane = threadIdx.x & 31;
    const int u0 = blockIdx.x * 16;          // unit tile
    const int b0 = blockIdx.y * 16;          // batch tile

    // --- asynchronously stage the contiguous 32 KB h tile into LDS ---------
    {
        const unsigned short* gsrc = h_prev + (size_t)b0 * UU;
#pragma unroll
        for (int i = 0; i < 16; ++i) {
            const int e = (i * 128 + (int)threadIdx.x) * 8;  // 16B per lane
            const unsigned short* gp = gsrc + e;
            const unsigned lp = (unsigned)(size_t)(hsm + e); // LDS addr[31:0]
            asm volatile("global_load_async_to_lds_b128 %0, %1, off"
                         :: "v"(lp), "v"(gp) : "memory");
        }
        asm volatile("s_wait_asynccnt 0x0" ::: "memory");
        __syncthreads();
    }

    v8f acc[4] = {};                          // i, f, g, o
    const int kbeg = wave * (UU / 4);
    const int kend = kbeg + (UU / 4);

    StepFrags cur, nxt;
    step_load_tile(cur, hsm, Rt, u0, kbeg, lane);
    for (int k0 = kbeg + 32; k0 < kend; k0 += 32) {
        step_load_tile(nxt, hsm, Rt, u0, k0, lane);       // overlap with WMMA
#pragma unroll
        for (int g = 0; g < 4; ++g)
            acc[g] = wmma_bf16(cur.a, cur.b[g], acc[g]);
        cur = nxt;
    }
#pragma unroll
    for (int g = 0; g < 4; ++g)
        acc[g] = wmma_bf16(cur.a, cur.b[g], acc[g]);

#pragma unroll
    for (int g = 0; g < 4; ++g)
#pragma unroll
        for (int v = 0; v < 8; ++v)
            part[((wave * 4 + g) * 8 + v) * 32 + lane] = acc[g][v];
    __syncthreads();

    // Gate epilogue distributed across all 4 waves: wave w handles
    // accumulator rows v = 2w and 2w+1 (each covers 32 (b,u) elements).
    const int col = lane & 15, hi = lane >> 4;
    const int u = u0 + col;
#pragma unroll
    for (int vv = 0; vv < 2; ++vv) {
        const int v = wave * 2 + vv;
        const int b = b0 + hi * 8 + v;
        float zi = 0.f, zf = 0.f, zg = 0.f, zo = 0.f;
#pragma unroll
        for (int w = 0; w < 4; ++w) {
            zi += part[((w * 4 + 0) * 8 + v) * 32 + lane];
            zf += part[((w * 4 + 1) * 8 + v) * 32 + lane];
            zg += part[((w * 4 + 2) * 8 + v) * 32 + lane];
            zo += part[((w * 4 + 3) * 8 + v) * 32 + lane];
        }
        const size_t xzb = ((size_t)b * TT + t) * (size_t)U4;
        zi += xz[xzb + 0 * UU + u];
        zf += xz[xzb + 1 * UU + u];
        zg += xz[xzb + 2 * UU + u];
        zo += xz[xzb + 3 * UU + u];

        const float ig = fast_sigmoid(zi);
        const float fg = fast_sigmoid(zf);
        const float gg = fast_tanh(zg);
        const float og = fast_sigmoid(zo);

        const size_t su = (size_t)b * UU + u;
        const float cn = fg * c_state[su] + ig * gg;
        const float hn = og * fast_tanh(cn);

        c_state[su] = cn;
        h_next[su]  = f32_to_bf16(hn);
        lstm_bf[((size_t)b * TT + t) * UU + u] = f32_to_bf16(hn);
        if (is_last) { h_last[su] = hn; c_last[su] = cn; }
    }
}

// ---------------------------------------------------------------------------
// Host-side orchestration
// ---------------------------------------------------------------------------
extern "C" void kernel_launch(void* const* d_in, const int* in_sizes, int n_in,
                              void* d_out, int out_size, void* d_ws, size_t ws_size,
                              hipStream_t stream) {
    (void)in_sizes; (void)n_in; (void)out_size; (void)ws_size;

    const float* x    = (const float*)d_in[0];   // [B,T,D]
    const float* h0   = (const float*)d_in[1];   // [B,U]
    const float* c0   = (const float*)d_in[2];   // [B,U]
    const float* Wk   = (const float*)d_in[3];   // [D,4U]
    const float* Wr   = (const float*)d_in[4];   // [U,4U]
    const float* bias = (const float*)d_in[5];   // [4U]
    const float* Wo   = (const float*)d_in[6];   // [U,M]
    const float* bo   = (const float*)d_in[7];   // [M]

    float* out    = (float*)d_out;                         // [B,T,M]
    float* h_last = out + (size_t)BB * TT * MM;            // [B,U]
    float* c_last = h_last + (size_t)BB * UU;              // [B,U]

    // workspace carve-out (256B aligned)
    char* p = (char*)d_ws;
    auto carve = [&](size_t bytes) {
        char* r = p; p += (bytes + 255) & ~(size_t)255; return r;
    };
    float*          xz      = (float*)carve((size_t)BB * TT * U4 * 4); // 512 MB
    unsigned short* x_bf    = (unsigned short*)carve((size_t)BB * TT * DD * 2);
    unsigned short* Kt      = (unsigned short*)carve((size_t)U4 * DD * 2);
    unsigned short* Rt      = (unsigned short*)carve((size_t)U4 * UU * 2);
    unsigned short* Wt      = (unsigned short*)carve((size_t)MM * UU * 2);
    unsigned short* lstm_bf = (unsigned short*)carve((size_t)BB * TT * UU * 2);
    unsigned short* hbuf0   = (unsigned short*)carve((size_t)BB * UU * 2);
    unsigned short* hbuf1   = (unsigned short*)carve((size_t)BB * UU * 2);
    float*          c_ws    = (float*)carve((size_t)BB * UU * 4);

    // --- phase 0: precision conversion + transposes -------------------------
    {
        int n = BB * TT * DD;
        k_convert_bf16<<<(n + 255) / 256, 256, 0, stream>>>(x, x_bf, n);
    }
    {
        int n = DD * U4;
        k_transpose_bf16<<<(n + 255) / 256, 256, 0, stream>>>(Wk, Kt, DD, U4);
    }
    {
        int n = UU * U4;
        k_transpose_bf16<<<(n + 255) / 256, 256, 0, stream>>>(Wr, Rt, UU, U4);
    }
    {
        int n = UU * MM;
        k_transpose_bf16<<<(n + 255) / 256, 256, 0, stream>>>(Wo, Wt, UU, MM);
    }
    {
        int n = BB * UU;
        k_init_state<<<(n + 255) / 256, 256, 0, stream>>>(h0, c0, hbuf0, c_ws, n);
    }

    // --- phase 1: xz = x @ kernel + bias  ([32768,256]x[256,4096]) ----------
    {
        dim3 grid(U4 / 64, (BB * TT) / 256);
        k_gemm_bf16<<<grid, 256, 0, stream>>>(x_bf, Kt, bias, xz,
                                              BB * TT, U4, DD);
    }

    // --- phase 2: 512 sequential fused LSTM steps ---------------------------
    {
        dim3 grid(UU / 16, BB / 16);
        for (int t = 0; t < TT; ++t) {
            const unsigned short* hp = (t & 1) ? hbuf1 : hbuf0;
            unsigned short*       hn = (t & 1) ? hbuf0 : hbuf1;
            k_lstm_step<<<grid, 128, 0, stream>>>(hp, Rt, xz, c_ws, hn,
                                                  lstm_bf, h_last, c_last,
                                                  t, (t == TT - 1) ? 1 : 0);
        }
    }

    // --- phase 3: output = lstm_out @ w_out + b_out -------------------------
    {
        dim3 grid(MM / 64, (BB * TT) / 256);
        k_gemm_bf16<<<grid, 256, 0, stream>>>(lstm_bf, Wt, bo, out,
                                              BB * TT, MM, UU);
    }
}
